// GCN_2_58385785422151
// MI455X (gfx1250) — matrix operand order repaired and checked by reading, not compiled, not run on hardware
//
#include <hip/hip_runtime.h>

// ---------------------------------------------------------------------------
// SemGCN forward on MI455X (gfx1250), compile-only target.
//
// Every sem_gconv is refactored to a single bf16 WMMA GEMM
//   Y = [D*X | offd@X] @ [W0;W1] + b   (f32 accumulation)
//  - adj_kernel   : per-layer softmax adjacency -> d[n], offd[n*n]
//  - packw_kernel : pack [W0;W1] (K=2*fin, zero-padded) into the WMMA
//                   B-fragment layout (lane-major, 16 contiguous halfs/lane)
//  - prep_kernel  : per-sample; BN+ReLU (+residual add, +contour/symmetry
//                   fuse with 512x10 -> 32x160 reshape), node-mix with
//                   (d, offd), emit bf16 GEMM operand Q [M, Kp]
//  - gemm_kernel  : 256x32 macro-tile / block, 8 waves * (32x32 wave tile,
//                   4 accumulators), K-step 32, v_wmma_f32_16x16x32_bf16;
//                   fused bias, BN sum/sumsq via LDS ds_add reduction + one
//                   global atomic per channel, fused sigmoid for output layer
//  - bnstats_kernel: sum/sumsq -> per-channel scale/shift for next layer
// ---------------------------------------------------------------------------

typedef __attribute__((ext_vector_type(16))) __bf16 v16bf;
typedef __attribute__((ext_vector_type(8)))  __bf16 v8bf;
typedef __attribute__((ext_vector_type(8)))  float  v8f;

#define PREP_RAW  0
#define PREP_BN   1
#define PREP_RES  2
#define PREP_FUSE 3

#define GF_STATS   1
#define GF_SIGMOID 2

// ------------------------- adjacency softmax -------------------------------
__global__ __launch_bounds__(256) void adj_kernel(const int* __restrict__ idx,
                                                  const float* __restrict__ e,
                                                  int ne, int n,
                                                  float* __restrict__ dvec,
                                                  float* __restrict__ offd) {
  __shared__ float lg[32 * 32];
  const int tid = threadIdx.x;
  for (int t = tid; t < n * n; t += 256) lg[t] = -9e15f;
  __syncthreads();
  for (int k = tid; k < ne; k += 256) lg[idx[2 * k] * n + idx[2 * k + 1]] = e[k];
  __syncthreads();
  if (tid < n) {
    float mx = -1e30f;
    for (int j = 0; j < n; ++j) mx = fmaxf(mx, lg[tid * n + j]);
    float s = 0.f;
    for (int j = 0; j < n; ++j) s += __expf(lg[tid * n + j] - mx);
    float inv = 1.f / s;
    for (int j = 0; j < n; ++j) {
      float a = __expf(lg[tid * n + j] - mx) * inv;
      offd[tid * n + j] = (j == tid) ? 0.f : a;
    }
    dvec[tid] = __expf(lg[tid * n + tid] - mx) * inv;
  }
}

// ------------------------- weight packing ----------------------------------
// PW[((n16*K32 + k32)*32 + lane)*16 + e] = Wcat[k][n], bf16
//   k = k32*32 + (lane>>4)*8 + (e&7) + (e>>3)*16 ,  n = n16*16 + (lane&15)
// Wcat = [W0 ; W1] along K, zero-padded to (Kp, Np).
__global__ __launch_bounds__(256) void packw_kernel(const float* __restrict__ W,
                                                    int fin, int fout, int Kp, int Np,
                                                    __bf16* __restrict__ PW) {
  const int k32t = Kp >> 5;
  const size_t total = (size_t)Np * (size_t)Kp;
  for (size_t t = (size_t)blockIdx.x * 256 + threadIdx.x; t < total;
       t += (size_t)gridDim.x * 256) {
    int el   = (int)(t & 15);
    int lane = (int)((t >> 4) & 31);
    size_t tile = t >> 9;
    int k32 = (int)(tile % (size_t)k32t);
    int n16 = (int)(tile / (size_t)k32t);
    int k  = k32 * 32 + ((lane >> 4) << 3) + (el & 7) + ((el >> 3) << 4);
    int nn = n16 * 16 + (lane & 15);
    float v = 0.f;
    if (nn < fout) {
      if (k < fin)            v = W[(size_t)k * fout + nn];                       // W0
      else if (k < 2 * fin)   v = W[(size_t)fin * fout + (size_t)(k - fin) * fout + nn]; // W1
    }
    PW[t] = (__bf16)v;
  }
}

// ------------------------- per-sample prep ---------------------------------
__global__ __launch_bounds__(256) void prep_kernel(
    const float* __restrict__ src0, const float* __restrict__ scsh0,
    const float* __restrict__ src1, const float* __restrict__ scsh1,
    const float* __restrict__ srcS, const float* __restrict__ scshS,
    const float* __restrict__ dvec, const float* __restrict__ offd,
    __bf16* __restrict__ Q, float* __restrict__ stats_zero,
    int n, int fin, int Kp, int mode, int zcount) {
  extern __shared__ float sm[];
  float* act  = sm;               // n*fin
  float* sd   = sm + n * fin;     // n
  float* soff = sd + n;           // n*n
  const int b = blockIdx.x;
  const int tid = threadIdx.x;

  if (b == 0 && stats_zero) {     // zero BN accumulators for this layer's GEMM
    for (int i = tid; i < zcount; i += 256) stats_zero[i] = 0.f;
  }
  for (int i = tid; i < n; i += 256) sd[i] = dvec[i];
  for (int i = tid; i < n * n; i += 256) soff[i] = offd[i];

  const int tot = n * fin;
  for (int t = tid; t < tot; t += 256) {
    int i = t / fin, c = t - i * fin;
    float v;
    if (mode == PREP_RAW) {
      v = src0[(size_t)b * tot + t];
    } else if (mode == PREP_BN) {
      float r = src0[(size_t)b * tot + t];
      v = fmaxf(r * scsh0[c] + scsh0[fin + c], 0.f);
    } else if (mode == PREP_RES) {
      float r0 = src0[(size_t)b * tot + t];
      float r1 = src1[(size_t)b * tot + t];
      v = fmaxf(r0 * scsh0[c] + scsh0[fin + c], 0.f) +
          fmaxf(r1 * scsh1[c] + scsh1[fin + c], 0.f);
    } else { // PREP_FUSE: fin==320, c-part (160 ch) + reshaped s-part (512 ch)
      if (c < 160) {
        size_t off = ((size_t)b * 32 + i) * 160 + c;
        float v0 = fmaxf(src0[off] * scsh0[c] + scsh0[160 + c], 0.f);
        float v1 = src1 ? fmaxf(src1[off] * scsh1[c] + scsh1[160 + c], 0.f) : 0.f;
        v = v0 + v1;
      } else {
        int tt = i * 160 + (c - 160);  // linear index into [512,10]
        int ch = tt / 10, j = tt - ch * 10;
        float r = srcS[((size_t)b * 10 + j) * 512 + ch];
        v = fmaxf(r * scshS[ch] + scshS[512 + ch], 0.f);
      }
    }
    act[t] = v;
  }
  __syncthreads();

  const size_t qbase = (size_t)b * n * Kp;
  const int totq = n * Kp;
  for (int t = tid; t < totq; t += 256) {
    int i = t / Kp, k = t - i * Kp;
    float v;
    if (k < fin) {
      v = sd[i] * act[i * fin + k];                // D * X
    } else if (k < 2 * fin) {
      int kk = k - fin;
      float s = 0.f;
      const float* orow = soff + i * n;
      const float* acol = act + kk;
      for (int j = 0; j < n; ++j) s += orow[j] * acol[j * fin];  // offd @ X
      v = s;
    } else {
      v = 0.f;                                     // K zero-pad
    }
    Q[qbase + t] = (__bf16)v;
  }
}

// ------------------------- WMMA GEMM ---------------------------------------
// Block: 256x32 macro-tile, 8 waves; wave tile 32x32 = 4 accumulators.
// Per K-step/wave: 2 A-frags + 2 B-frags (4 KB) feed 4 WMMAs (16 flop/byte).
__global__ __launch_bounds__(256) void gemm_kernel(
    const __bf16* __restrict__ Q, const __bf16* __restrict__ PW,
    const float* __restrict__ bias, float* __restrict__ Y,
    float* __restrict__ stats, int M, int Kp, int fout, int Np, int flags) {
  const int lane = threadIdx.x & 31;
  const int wave = threadIdx.x >> 5;
  const int m0 = blockIdx.x * 256 + wave * 32;
  const int n0 = blockIdx.y * 32;
  const int k32t = Kp >> 5;

  int rA = m0 + (lane & 15);        if (rA >= M) rA = M - 1;
  int rB = m0 + 16 + (lane & 15);   if (rB >= M) rB = M - 1;
  const int kb = (lane >> 4) * 8;
  const __bf16* qrow0 = Q + (size_t)rA * Kp + kb;
  const __bf16* qrow1 = Q + (size_t)rB * Kp + kb;
  const __bf16* pw0 = PW + ((size_t)(n0 >> 4) * k32t) * 512 + (size_t)lane * 16;
  const __bf16* pw1 = PW + ((size_t)((n0 >> 4) + 1) * k32t) * 512 + (size_t)lane * 16;

  v8f acc00 = {}, acc01 = {}, acc10 = {}, acc11 = {};
  for (int k = 0; k < k32t; ++k) {
    __builtin_prefetch((const void*)(qrow0 + 256), 0, 3);  // near-cache prefetch
    __builtin_prefetch((const void*)(qrow1 + 256), 0, 3);
    v8bf lo0 = *(const v8bf*)(qrow0);
    v8bf hi0 = *(const v8bf*)(qrow0 + 16);
    v8bf lo1 = *(const v8bf*)(qrow1);
    v8bf hi1 = *(const v8bf*)(qrow1 + 16);
    qrow0 += 32; qrow1 += 32;
    v16bf a0, a1;
#pragma unroll
    for (int e = 0; e < 8; ++e) {
      a0[e] = lo0[e]; a0[e + 8] = hi0[e];
      a1[e] = lo1[e]; a1[e + 8] = hi1[e];
    }
    v16bf b0 = *(const v16bf*)(pw0); pw0 += 512;
    v16bf b1 = *(const v16bf*)(pw1); pw1 += 512;
    acc00 = __builtin_amdgcn_wmma_f32_16x16x32_bf16(false, a0, false, b0,
                                                    (short)0, acc00, false, false);
    acc01 = __builtin_amdgcn_wmma_f32_16x16x32_bf16(false, a0, false, b1,
                                                    (short)0, acc01, false, false);
    acc10 = __builtin_amdgcn_wmma_f32_16x16x32_bf16(false, a1, false, b0,
                                                    (short)0, acc10, false, false);
    acc11 = __builtin_amdgcn_wmma_f32_16x16x32_bf16(false, a1, false, b1,
                                                    (short)0, acc11, false, false);
  }

  __shared__ float ssum[32], ssq[32];
  if (threadIdx.x < 32) { ssum[threadIdx.x] = 0.f; ssq[threadIdx.x] = 0.f; }
  __syncthreads();

#pragma unroll
  for (int mt = 0; mt < 2; ++mt) {
    const int mrow = m0 + mt * 16 + ((lane >> 4) << 3);
#pragma unroll
    for (int t = 0; t < 2; ++t) {
      const int col = n0 + t * 16 + (lane & 15);
      v8f acc = mt ? (t ? acc11 : acc10) : (t ? acc01 : acc00);
      float lsum = 0.f, lsq = 0.f;
      if (col < fout) {
        const float bv = bias ? bias[col] : 0.f;
#pragma unroll
        for (int v = 0; v < 8; ++v) {
          int mr = mrow + v;
          float val = acc[v] + bv;
          if (flags & GF_SIGMOID) val = 1.f / (1.f + __expf(-val));
          if (mr < M) {
            Y[(size_t)mr * fout + col] = val;
            lsum += val; lsq += val * val;
          }
        }
        if (flags & GF_STATS) {                     // LDS ds_add reduction
          atomicAdd(&ssum[t * 16 + (lane & 15)], lsum);
          atomicAdd(&ssq [t * 16 + (lane & 15)], lsq);
        }
      }
    }
  }
  __syncthreads();
  if ((flags & GF_STATS) && threadIdx.x < 32) {
    int col = n0 + threadIdx.x;
    if (col < fout) {
      atomicAdd(&stats[col],        ssum[threadIdx.x]);
      atomicAdd(&stats[fout + col], ssq[threadIdx.x]);
    }
  }
}

// ------------------------- BN stats -> scale/shift -------------------------
__global__ void bnstats_kernel(const float* __restrict__ stats,
                               const float* __restrict__ gamma,
                               const float* __restrict__ beta,
                               float* __restrict__ scsh, int fout, float invcnt) {
  int c = blockIdx.x * blockDim.x + threadIdx.x;
  if (c < fout) {
    float mu  = stats[c] * invcnt;
    float var = stats[fout + c] * invcnt - mu * mu;
    float s = gamma[c] * rsqrtf(var + 1e-5f);
    scsh[c] = s;
    scsh[fout + c] = beta[c] - mu * s;
  }
}

// ===========================================================================
extern "C" void kernel_launch(void* const* d_in, const int* in_sizes, int n_in,
                              void* d_out, int out_size, void* d_ws, size_t ws_size,
                              hipStream_t stream) {
  (void)n_in; (void)ws_size; (void)out_size;
  const float* x_c = (const float*)d_in[0];
  const float* x_s = (const float*)d_in[1];
  const int*   idx_c = (const int*)d_in[70];
  const int*   idx_s = (const int*)d_in[71];
  const int B = in_sizes[0] / 64;         // x_c is [B,32,2]

  // layer order: gin,s1,c1a,c1b,c1c,s2,c2a,c2b,c2c,s3,c3a,c3b,c3c,gout
  static const int FIN [14] = {  2,  2,320,160,160,512,320,160,160,512,320,160,160,160};
  static const int FOUT[14] = {160,512,160,160,160,512,160,160,160,512,160,160,160,  2};
  static const int NND [14] = { 32, 10, 32, 32, 32, 10, 32, 32, 32, 10, 32, 32, 32, 32};

  struct LayerP {
    const float *W, *e, *b, *gamma, *beta;
    int fin, fout, n, ne, Kp, Np;
    float *dv, *od, *scsh, *stats;
    __bf16* pw;
  } L[14];

  for (int i = 0; i < 14; ++i) {
    if (i < 13) {
      int base = 2 + 5 * i;
      L[i].W = (const float*)d_in[base];     L[i].e = (const float*)d_in[base + 1];
      L[i].b = (const float*)d_in[base + 2]; L[i].gamma = (const float*)d_in[base + 3];
      L[i].beta = (const float*)d_in[base + 4];
      L[i].ne = in_sizes[base + 1];
    } else {
      L[i].W = (const float*)d_in[67]; L[i].e = (const float*)d_in[68];
      L[i].b = (const float*)d_in[69]; L[i].gamma = nullptr; L[i].beta = nullptr;
      L[i].ne = in_sizes[68];
    }
    L[i].fin = FIN[i]; L[i].fout = FOUT[i]; L[i].n = NND[i];
    int k2 = 2 * FIN[i];
    L[i].Kp = (k2 < 32) ? 32 : ((k2 + 31) & ~31);
    L[i].Np = (FOUT[i] + 31) & ~31;
  }

  // ---- workspace bump allocator --------------------------------------------
  size_t off = 0;
  auto alloc = [&](size_t bytes) -> void* {
    void* p = (char*)d_ws + off;
    off = (off + bytes + 255) & ~(size_t)255;
    return p;
  };
  for (int i = 0; i < 14; ++i) {
    L[i].dv = (float*)alloc(32 * sizeof(float));
    L[i].od = (float*)alloc(32 * 32 * sizeof(float));
    L[i].scsh  = (float*)alloc(1024 * sizeof(float));
    L[i].stats = (float*)alloc(1024 * sizeof(float));
    L[i].pw = (__bf16*)alloc((size_t)L[i].Kp * L[i].Np * sizeof(__bf16));
  }
  const size_t szY = (size_t)B * 32 * 160 * sizeof(float);   // == B*10*512*4
  float* YA = (float*)alloc(szY);
  float* YB = (float*)alloc(szY);
  float* YC = (float*)alloc(szY);
  float* S1 = (float*)alloc(szY);
  float* S2 = (float*)alloc(szY);
  __bf16* Qbuf = (__bf16*)alloc((size_t)B * 32 * 640 * sizeof(__bf16)); // max [M,Kp]

  // ---- per-launch constant prep: adjacency softmax + weight packing --------
  for (int i = 0; i < 14; ++i) {
    const int* idx = (L[i].n == 32) ? idx_c : idx_s;
    adj_kernel<<<1, 256, 0, stream>>>(idx, L[i].e, L[i].ne, L[i].n, L[i].dv, L[i].od);
    size_t total = (size_t)L[i].Np * L[i].Kp;
    int blocks = (int)((total + 255) / 256);
    packw_kernel<<<blocks, 256, 0, stream>>>(L[i].W, L[i].fin, L[i].fout,
                                             L[i].Kp, L[i].Np, L[i].pw);
  }

  // ---- layer launch helpers ------------------------------------------------
  auto prep = [&](int li, int mode, const float* s0, const float* sc0,
                  const float* s1p, const float* sc1,
                  const float* sS, const float* scS, bool zero_stats) {
    const LayerP& P = L[li];
    size_t shb = (size_t)(P.n * P.fin + P.n + P.n * P.n) * sizeof(float);
    prep_kernel<<<B, 256, shb, stream>>>(s0, sc0, s1p, sc1, sS, scS,
                                         P.dv, P.od, Qbuf,
                                         zero_stats ? P.stats : nullptr,
                                         P.n, P.fin, P.Kp, mode, 2 * P.fout);
  };
  auto gemm = [&](int li, float* Yout, int flags) {
    const LayerP& P = L[li];
    int M = B * P.n;
    dim3 g((M + 255) / 256, P.Np / 32);
    gemm_kernel<<<g, 256, 0, stream>>>(Qbuf, P.pw, P.b, Yout, P.stats,
                                       M, P.Kp, P.fout, P.Np, flags);
  };
  auto bnstats = [&](int li) {
    const LayerP& P = L[li];
    bnstats_kernel<<<(P.fout + 255) / 256, 256, 0, stream>>>(
        P.stats, P.gamma, P.beta, P.scsh, P.fout, 1.f / (float)(B * P.n));
  };

  // ---- network ------------------------------------------------------------
  // gin: c0 = gc(x_c)         -> YA
  prep(0, PREP_RAW, x_c, nullptr, nullptr, nullptr, nullptr, nullptr, true);
  gemm(0, YA, GF_STATS); bnstats(0);
  // s1: s0 = gc(x_s)          -> S1
  prep(1, PREP_RAW, x_s, nullptr, nullptr, nullptr, nullptr, nullptr, true);
  gemm(1, S1, GF_STATS); bnstats(1);

  // block 1
  prep(2, PREP_FUSE, YA, L[0].scsh, nullptr, nullptr, S1, L[1].scsh, true);  // fuse(c0,s0)
  gemm(2, YB, GF_STATS); bnstats(2);                                          // Yc1a -> YB
  prep(3, PREP_BN, YB, L[2].scsh, nullptr, nullptr, nullptr, nullptr, true);
  gemm(3, YC, GF_STATS); bnstats(3);                                          // Yc1b -> YC
  prep(4, PREP_BN, YC, L[3].scsh, nullptr, nullptr, nullptr, nullptr, true);
  gemm(4, YA, GF_STATS); bnstats(4);                                          // Yc1c -> YA
  prep(5, PREP_BN, S1, L[1].scsh, nullptr, nullptr, nullptr, nullptr, true);
  gemm(5, S2, GF_STATS); bnstats(5);                                          // Ys2  -> S2

  // block 2: c1' = bnrelu(YB)+bnrelu(YA)
  prep(6, PREP_FUSE, YB, L[2].scsh, YA, L[4].scsh, S2, L[5].scsh, true);
  gemm(6, YC, GF_STATS); bnstats(6);                                          // Yc2a -> YC
  prep(7, PREP_BN, YC, L[6].scsh, nullptr, nullptr, nullptr, nullptr, true);
  gemm(7, YB, GF_STATS); bnstats(7);                                          // Yc2b -> YB
  prep(8, PREP_BN, YB, L[7].scsh, nullptr, nullptr, nullptr, nullptr, true);
  gemm(8, YA, GF_STATS); bnstats(8);                                          // Yc2c -> YA
  prep(9, PREP_BN, S2, L[5].scsh, nullptr, nullptr, nullptr, nullptr, true);
  gemm(9, S1, GF_STATS); bnstats(9);                                          // Ys3  -> S1

  // block 3: c2' = bnrelu(YC)+bnrelu(YA)
  prep(10, PREP_FUSE, YC, L[6].scsh, YA, L[8].scsh, S1, L[9].scsh, true);
  gemm(10, YB, GF_STATS); bnstats(10);                                        // Yc3a -> YB
  prep(11, PREP_BN, YB, L[10].scsh, nullptr, nullptr, nullptr, nullptr, true);
  gemm(11, YC, GF_STATS); bnstats(11);                                        // Yc3b -> YC
  prep(12, PREP_BN, YC, L[11].scsh, nullptr, nullptr, nullptr, nullptr, true);
  gemm(12, YA, GF_STATS); bnstats(12);                                        // Yc3c -> YA

  // gout: sigmoid(sem_gconv(c3')) ; c3' = bnrelu(YB)+bnrelu(YA)
  prep(13, PREP_RES, YB, L[10].scsh, YA, L[12].scsh, nullptr, nullptr, false);
  gemm(13, (float*)d_out, GF_SIGMOID);
}
